// TalkingHeads_32796370272362
// MI455X (gfx1250) — compile-verified, matrix operand order checked by reading, not run on hardware
//
#include <hip/hip_runtime.h>
#include <hip/hip_bf16.h>

typedef __bf16 bhalf;
typedef __attribute__((ext_vector_type(16))) __bf16 v16bf;
typedef __attribute__((ext_vector_type(8)))  float  v8f;
typedef __attribute__((ext_vector_type(2)))  float  v2f;
typedef __attribute__((ext_vector_type(4)))  unsigned int v4u;
typedef __attribute__((ext_vector_type(8)))  int    v8i;
typedef __attribute__((ext_vector_type(4)))  int    v4i;

#define SEQL 1024
#define CIN  512
#define DHEAD 64
#define NH   16

__device__ __forceinline__ unsigned rfl_u(unsigned x) {
  return (unsigned)__builtin_amdgcn_readfirstlane((int)x);
}

// Tensor Data Mover: 2D tile load (rows x cols bf16, row stride in elems)
// into LDS at byte offset lds_off.  6-arg builtin (clang-23 form); D# per ISA §8.
__device__ __forceinline__ void tdm_load_2d_bf16(const bhalf* gptr, unsigned lds_off,
                                                 int rows, int cols, int row_stride) {
  unsigned long long ga = (unsigned long long)(uintptr_t)gptr;
  const unsigned ga_lo = rfl_u((unsigned)ga);
  const unsigned ga_hi = rfl_u((unsigned)(ga >> 32));
  v4u g0;
  g0[0] = 1u;                                            // count=1, user mode
  g0[1] = rfl_u(lds_off);                                // lds_addr (bytes)
  g0[2] = ga_lo;                                         // global_addr[31:0]
  g0[3] = (ga_hi & 0x01FFFFFFu) | (2u << 30);            // addr[56:32] | type=2
  v8i g1;
  g1[0] = (int)(1u << 16);                               // data_size=1 (2B), mask=0
  g1[1] = (int)(((unsigned)cols & 0xFFFFu) << 16);       // tensor_dim0[15:0]
  g1[2] = (int)((((unsigned)cols >> 16) & 0xFFFFu) |
                (((unsigned)rows & 0xFFFFu) << 16));     // t_dim0 hi | t_dim1 lo
  g1[3] = (int)((((unsigned)rows >> 16) & 0xFFFFu) |
                (((unsigned)cols & 0xFFFFu) << 16));     // t_dim1 hi | tile_dim0
  g1[4] = (int)((unsigned)rows & 0xFFFFu);               // tile_dim1 | tile_dim2=0
  g1[5] = (int)(unsigned)row_stride;                     // tensor_dim0_stride lo32
  g1[6] = 0;                                             // stride hi | dim1_stride
  g1[7] = 0;
  v4i z4 = {0, 0, 0, 0};
  v8i z8 = {0, 0, 0, 0, 0, 0, 0, 0};
  __builtin_amdgcn_tensor_load_to_lds(g0, g1, z4, z4, z8, 0);
}

// ---------------------------------------------------------------------------
// Kernel 1: Q/K/V projections.  Y[s, n] = sum_c inp[b][c][s] * W[c][n],
// n = d*16 + h; scaled by sigmoid(alpha); stored bf16 as [b][h][s][d].
// grid: x = 64 s-tiles, y = 8 n-groups(128), z = b*3 + which
// ---------------------------------------------------------------------------
__global__ __launch_bounds__(256) void th_proj_qkv(
    const float* __restrict__ inp,
    const float* __restrict__ Wq, const float* __restrict__ Wk, const float* __restrict__ Wv,
    const float* __restrict__ aq, const float* __restrict__ ak, const float* __restrict__ av,
    bhalf* __restrict__ Qws, bhalf* __restrict__ Kws, bhalf* __restrict__ Vws)
{
  const int lane = threadIdx.x & 31;
  const int wave = threadIdx.x >> 5;
  const int s0 = blockIdx.x * 16;
  const int n0 = blockIdx.y * 128 + wave * 16;
  const int b  = blockIdx.z / 3;
  const int w  = blockIdx.z % 3;
  const float* Wm  = (w == 0) ? Wq : (w == 1) ? Wk : Wv;
  const float* al  = (w == 0) ? aq : (w == 1) ? ak : av;
  bhalf*       dst = (w == 0) ? Qws : (w == 1) ? Kws : Vws;
  const float sig = 1.f / (1.f + __expf(-al[0]));

  const int m    = lane & 15;
  const int half = lane >> 4;
  const int kb   = half * 8;
  const float* inpB = inp + (size_t)b * CIN * SEQL;

  v8f acc = {};
  for (int c0 = 0; c0 < CIN; c0 += 32) {
    v16bf a, bf;
#pragma unroll
    for (int j = 0; j < 8; ++j) {
      const int kA = c0 + kb + j;
      const int kB = c0 + 16 + kb + j;
      a[j]    = (bhalf)inpB[(size_t)kA * SEQL + s0 + m];
      a[j+8]  = (bhalf)inpB[(size_t)kB * SEQL + s0 + m];
      bf[j]   = (bhalf)Wm[(size_t)kA * 1024 + n0 + m];
      bf[j+8] = (bhalf)Wm[(size_t)kB * 1024 + n0 + m];
    }
    acc = __builtin_amdgcn_wmma_f32_16x16x32_bf16(false, a, false, bf,
                                                  (short)0, acc, false, false);
  }
  const int n = n0 + m;
  const int h = n & 15, d = n >> 4;
  bhalf* o = dst + ((size_t)(b * NH + h) * SEQL) * DHEAD + d;
#pragma unroll
  for (int v = 0; v < 8; ++v) {
    const int s = s0 + v + half * 8;
    o[(size_t)s * DHEAD] = (bhalf)(sig * acc[v]);
  }
}

// ---------------------------------------------------------------------------
// Kernel 2: fused talking-heads attention, flash-style, two passes over K.
// 16 waves/block; wave w = head h (QK^T via TDM-staged K), q-row w (mix +
// softmax), value head w (U*V via TDM-staged V).  K and V share one LDS
// staging tile per head, guarded by s_wait_dscnt / s_wait_tensorcnt.
// grid: x = 64 q-tiles, y = b
// ---------------------------------------------------------------------------
__global__ __launch_bounds__(512) void th_attn(
    const bhalf* __restrict__ Qws, const bhalf* __restrict__ Kws,
    const bhalf* __restrict__ Vws,
    const float* __restrict__ Wl, const float* __restrict__ Ww,
    const float* __restrict__ mask, bhalf* __restrict__ Ows)
{
  __shared__ bhalf Tlds[NH][32][DHEAD];  // K/V staging tile per head  64 KB
  __shared__ float Jlds[NH][16][32];     // [h][q][k]                  32 KB
  __shared__ float Wlds[16][32][NH];     // [q][k][g]                  32 KB
  __shared__ bhalf Ulds[16][32][NH];     // [q][k][v]                  16 KB

  const int lane = threadIdx.x & 31;
  const int wave = threadIdx.x >> 5;     // 0..15
  const int b    = blockIdx.y;
  const int q0   = blockIdx.x * 16;
  const int m    = lane & 15;
  const int half = lane >> 4;
  const int kb   = half * 8;

  const bhalf* Qh = Qws + ((size_t)(b * NH + wave) * SEQL) * DHEAD;
  const bhalf* Kh = Kws + ((size_t)(b * NH + wave) * SEQL) * DHEAD;
  const bhalf* Vh = Vws + ((size_t)(b * NH + wave) * SEQL) * DHEAD;
  const float* maskB = mask + (size_t)b * SEQL;
  const unsigned tl_off = (unsigned)(uintptr_t)(&Tlds[wave][0][0]);

  // persistent Q fragments for head h = wave (A of 16x32, d-chunks 0/1)
  v16bf qa[2];
#pragma unroll
  for (int c = 0; c < 2; ++c)
#pragma unroll
    for (int j = 0; j < 8; ++j) {
      qa[c][j]   = Qh[(size_t)(q0 + m) * DHEAD + c * 32 + kb + j];
      qa[c][j+8] = Qh[(size_t)(q0 + m) * DHEAD + c * 32 + 16 + kb + j];
    }

  float mg = -1e30f, lg = 0.f;           // softmax stats (q=wave, g=lane&15)
  v8f oacc[4] = {{}, {}, {}, {}};        // O for v=wave: 16(q) x 64(d)

  for (int pass = 0; pass < 2; ++pass) {
    for (int kt = 0; kt < SEQL; kt += 32) {
      // ---- stage this head's K tile (32k x 64d) via Tensor Data Mover
      asm volatile("s_wait_dscnt 0x0" ::: "memory");       // prior tile reads done
      tdm_load_2d_bf16(Kh + (size_t)kt * DHEAD, tl_off, 32, DHEAD, DHEAD);
      __builtin_amdgcn_s_wait_tensorcnt(0);

      // ---- J = Q_h K_h^T : 16(q) x 32(k), K=64 (2 chunks)
#pragma unroll
      for (int t = 0; t < 2; ++t) {
        v8f jacc = {};
#pragma unroll
        for (int c = 0; c < 2; ++c) {
          v16bf kf;                      // B(d, n=k) = Tlds[wave][k][d]
#pragma unroll
          for (int j = 0; j < 8; ++j) {
            kf[j]   = Tlds[wave][t*16 + m][c*32 + kb + j];
            kf[j+8] = Tlds[wave][t*16 + m][c*32 + 16 + kb + j];
          }
          jacc = __builtin_amdgcn_wmma_f32_16x16x32_bf16(false, qa[c], false, kf,
                                                         (short)0, jacc, false, false);
        }
#pragma unroll
        for (int v = 0; v < 8; ++v)
          Jlds[wave][v + kb][t*16 + m] = jacc[v];
      }

      // K tile consumed -> asynchronously pull V tile into same buffer
      if (pass == 1) {
        asm volatile("s_wait_dscnt 0x0" ::: "memory");     // K-frag LDS reads done
        tdm_load_2d_bf16(Vh + (size_t)kt * DHEAD, tl_off, 32, DHEAD, DHEAD);
      }
      __syncthreads();

      // ---- EL = J x Wl for q = wave: rows k (32), cols g (16), K = h (16)
      v8f el[2];
#pragma unroll
      for (int t = 0; t < 2; ++t) {
        v8f e = {};
#pragma unroll
        for (int hc = 0; hc < 4; ++hc) {
          const int h0 = hc * 4 + half * 2;
          v2f a2, b2;
          a2[0] = Jlds[h0 + 0][wave][t*16 + m];
          a2[1] = Jlds[h0 + 1][wave][t*16 + m];
          b2[0] = Wl[(h0 + 0) * NH + m];
          b2[1] = Wl[(h0 + 1) * NH + m];
          e = __builtin_amdgcn_wmma_f32_16x16x4_f32(false, a2, false, b2,
                                                    (short)0, e, false, false);
        }
#pragma unroll
        for (int v = 0; v < 8; ++v)
          e[v] -= maskB[kt + t*16 + v + kb];
        el[t] = e;
      }

      if (pass == 0) {
        float tmax = -1e30f;
#pragma unroll
        for (int t = 0; t < 2; ++t)
#pragma unroll
          for (int v = 0; v < 8; ++v) tmax = fmaxf(tmax, el[t][v]);
        tmax = fmaxf(tmax, __shfl_xor(tmax, 16, 32));
        const float mnew = fmaxf(mg, tmax);
        float s = 0.f;
#pragma unroll
        for (int t = 0; t < 2; ++t)
#pragma unroll
          for (int v = 0; v < 8; ++v) s += __expf(el[t][v] - mnew);
        s += __shfl_xor(s, 16, 32);
        lg = lg * __expf(mg - mnew) + s;
        mg = mnew;
        __syncthreads();
        continue;
      }

      // ---- exact softmax weights
      const float invl = 1.f / lg;
#pragma unroll
      for (int t = 0; t < 2; ++t)
#pragma unroll
        for (int v = 0; v < 8; ++v)
          Wlds[wave][t*16 + v + kb][m] = __expf(el[t][v] - mg) * invl;

      // ---- U = Wn x Ww : rows k (32), K = g (16), cols v (16)
#pragma unroll
      for (int t = 0; t < 2; ++t) {
        v8f u = {};
#pragma unroll
        for (int gc = 0; gc < 4; ++gc) {
          const int g0 = gc * 4 + half * 2;
          v2f a2, b2;
          a2[0] = Wlds[wave][t*16 + m][g0 + 0];
          a2[1] = Wlds[wave][t*16 + m][g0 + 1];
          b2[0] = Ww[(g0 + 0) * NH + m];
          b2[1] = Ww[(g0 + 1) * NH + m];
          u = __builtin_amdgcn_wmma_f32_16x16x4_f32(false, a2, false, b2,
                                                    (short)0, u, false, false);
        }
#pragma unroll
        for (int v = 0; v < 8; ++v)
          Ulds[wave][t*16 + v + kb][m] = (bhalf)u[v];
      }
      __syncthreads();

      // ---- O_v += U_v x V_v (V tile now resident in Tlds[wave])
      __builtin_amdgcn_s_wait_tensorcnt(0);
      v16bf ua;
#pragma unroll
      for (int j = 0; j < 8; ++j) {
        ua[j]   = Ulds[m][kb + j][wave];
        ua[j+8] = Ulds[m][16 + kb + j][wave];
      }
#pragma unroll
      for (int dt = 0; dt < 4; ++dt) {
        v16bf vb;
#pragma unroll
        for (int j = 0; j < 8; ++j) {
          vb[j]   = Tlds[wave][kb + j][dt*16 + m];
          vb[j+8] = Tlds[wave][16 + kb + j][dt*16 + m];
        }
        oacc[dt] = __builtin_amdgcn_wmma_f32_16x16x32_bf16(false, ua, false, vb,
                                                           (short)0, oacc[dt], false, false);
      }
      __syncthreads();
    }
  }

  // ---- store O[b][q][c], c = d*16 + v  (v = wave)
  bhalf* Ob = Ows + ((size_t)b * SEQL + q0) * (DHEAD * NH);
#pragma unroll
  for (int dt = 0; dt < 4; ++dt) {
    const int d = dt * 16 + m;
    const int c = d * NH + wave;
#pragma unroll
    for (int v = 0; v < 8; ++v)
      Ob[(size_t)(v + kb) * (DHEAD * NH) + c] = (bhalf)oacc[dt][v];
  }
}

// ---------------------------------------------------------------------------
// Kernel 3: out[b][n][q] = inp[b][n][q] + sum_c O[b][q][c] * Wo[c][n]
// grid: x = 64 q-tiles, y = 4 n-groups(128), z = b
// ---------------------------------------------------------------------------
__global__ __launch_bounds__(256) void th_outproj(
    const bhalf* __restrict__ Ows, const float* __restrict__ Wo,
    const float* __restrict__ inp, float* __restrict__ out)
{
  const int lane = threadIdx.x & 31;
  const int wave = threadIdx.x >> 5;
  const int q0 = blockIdx.x * 16;
  const int n0 = blockIdx.y * 128 + wave * 16;
  const int b  = blockIdx.z;
  const int m    = lane & 15;
  const int half = lane >> 4;
  const int kb   = half * 8;
  const bhalf* Ob = Ows + (size_t)b * SEQL * (DHEAD * NH);

  v8f acc = {};
  for (int c0 = 0; c0 < DHEAD * NH; c0 += 32) {
    v16bf a, bf;
#pragma unroll
    for (int j = 0; j < 8; ++j) {
      const int kA = c0 + kb + j;
      const int kB = c0 + 16 + kb + j;
      a[j]    = Ob[(size_t)(q0 + m) * (DHEAD * NH) + kA];
      a[j+8]  = Ob[(size_t)(q0 + m) * (DHEAD * NH) + kB];
      bf[j]   = (bhalf)Wo[(size_t)kA * CIN + n0 + m];
      bf[j+8] = (bhalf)Wo[(size_t)kB * CIN + n0 + m];
    }
    acc = __builtin_amdgcn_wmma_f32_16x16x32_bf16(false, a, false, bf,
                                                  (short)0, acc, false, false);
  }
  const int n = n0 + m;
  const size_t base = (size_t)b * CIN * SEQL + (size_t)n * SEQL + q0;
#pragma unroll
  for (int v = 0; v < 8; ++v) {
    const int q = v + half * 8;
    out[base + q] = inp[base + q] + acc[v];
  }
}

// ---------------------------------------------------------------------------
extern "C" void kernel_launch(void* const* d_in, const int* in_sizes, int n_in,
                              void* d_out, int out_size, void* d_ws, size_t ws_size,
                              hipStream_t stream) {
  (void)in_sizes; (void)n_in; (void)out_size; (void)ws_size;
  const float* inp  = (const float*)d_in[0];
  const float* mask = (const float*)d_in[1];
  const float* Wq   = (const float*)d_in[2];
  const float* Wk   = (const float*)d_in[3];
  const float* Wv   = (const float*)d_in[4];
  const float* aq   = (const float*)d_in[5];
  const float* ak   = (const float*)d_in[6];
  const float* av   = (const float*)d_in[7];
  const float* Wl   = (const float*)d_in[8];
  const float* Ww   = (const float*)d_in[9];
  const float* Wo   = (const float*)d_in[10];
  float* out = (float*)d_out;

  const size_t qkv_elems = (size_t)2 * NH * SEQL * DHEAD;
  bhalf* Qws = (bhalf*)d_ws;
  bhalf* Kws = Qws + qkv_elems;
  bhalf* Vws = Kws + qkv_elems;
  bhalf* Ows = Vws + qkv_elems;

  th_proj_qkv<<<dim3(64, 8, 6), 256, 0, stream>>>(inp, Wq, Wk, Wv, aq, ak, av,
                                                  Qws, Kws, Vws);
  th_attn<<<dim3(64, 2), 512, 0, stream>>>(Qws, Kws, Vws, Wl, Ww, mask, Ows);
  th_outproj<<<dim3(64, 4, 2), 256, 0, stream>>>(Ows, Wo, inp, out);
}
